// DistanceBiasedTemporalAttention_944892805293
// MI455X (gfx1250) — matrix operand order repaired and checked by reading, not compiled, not run on hardware
//
#include <hip/hip_runtime.h>
#include <hip/hip_bf16.h>

// ---------------------------------------------------------------------------
// DistanceBiasedTemporalAttention for MI455X (gfx1250, wave32, WMMA bf16)
// P=8192, D=512.  All big GEMMs via v_wmma_f32_16x16x32_bf16.
// ---------------------------------------------------------------------------

#define P_DIM 8192
#define D_DIM 512
#define SCALE_QK 0.044194173824159216f   // 1/sqrt(512)

typedef __attribute__((ext_vector_type(16))) __bf16 v16bf;
typedef __attribute__((ext_vector_type(8)))  __bf16 bf16x8;
typedef __attribute__((ext_vector_type(8)))  float  v8f;

// ---- WMMA fragment loaders (ISA 7.12.2 layouts, wave32) -------------------
__device__ __forceinline__ v16bf load_frag_a(const __bf16* base, int ld,
                                             int r0, int k0, int lane) {
    const int m  = lane & 15;
    const int hi = lane >> 4;
    const __bf16* p = base + (size_t)(r0 + m) * ld + k0 + hi * 8;
    bf16x8 lo = *(const bf16x8*)p;
    bf16x8 h8 = *(const bf16x8*)(p + 16);
    return __builtin_shufflevector(lo, h8, 0,1,2,3,4,5,6,7,8,9,10,11,12,13,14,15);
}

__device__ __forceinline__ v16bf load_frag_b(const __bf16* base, int ld,
                                             int n0, int k0, int lane) {
    const int n  = lane & 15;
    const int hi = lane >> 4;
    const __bf16* p = base + (size_t)(n0 + n) * ld + k0 + hi * 8;
    bf16x8 lo = *(const bf16x8*)p;
    bf16x8 h8 = *(const bf16x8*)(p + 16);
    return __builtin_shufflevector(lo, h8, 0,1,2,3,4,5,6,7,8,9,10,11,12,13,14,15);
}

__device__ __forceinline__ v8f wmma_bf16(v16bf a, v16bf b, v8f c) {
    return __builtin_amdgcn_wmma_f32_16x16x32_bf16(
        false, a, false, b, (short)0, c, false, false);
}

// ---------------------------------------------------------------------------
// Kernel 1: F (f32) -> Fb (bf16), sqn[p] = sum_d F[p,d]^2.  One wave per row.
// ---------------------------------------------------------------------------
__global__ void k_prep_rows(const float* __restrict__ F,
                            __bf16* __restrict__ Fb,
                            float* __restrict__ sqn) {
    const int wave = threadIdx.x >> 5;
    const int lane = threadIdx.x & 31;
    const int row  = blockIdx.x * 8 + wave;
    if (row >= P_DIM) return;
    const float* src = F + (size_t)row * D_DIM;
    __bf16*      dst = Fb + (size_t)row * D_DIM;
    float acc = 0.f;
    #pragma unroll
    for (int j = 0; j < 4; ++j) {
        const int base = j * 128 + lane * 4;
        float4 x = *(const float4*)(src + base);
        acc += x.x * x.x + x.y * x.y + x.z * x.z + x.w * x.w;
        dst[base + 0] = (__bf16)x.x;
        dst[base + 1] = (__bf16)x.y;
        dst[base + 2] = (__bf16)x.z;
        dst[base + 3] = (__bf16)x.w;
    }
    #pragma unroll
    for (int mask = 16; mask >= 1; mask >>= 1)
        acc += __shfl_xor(acc, mask, 32);
    if (lane == 0) sqn[row] = acc;
}

// ---------------------------------------------------------------------------
// Kernel 2: Wq/Wk/Wv (f32, [D][D]) -> Wb (bf16, [3][D][D])
// ---------------------------------------------------------------------------
__global__ void k_prep_w(const float* __restrict__ wq,
                         const float* __restrict__ wk,
                         const float* __restrict__ wv,
                         __bf16* __restrict__ Wb) {
    const int idx = blockIdx.x * blockDim.x + threadIdx.x;   // < 3*512*512
    const int mat = idx >> 18;
    const int off = idx & ((1 << 18) - 1);
    const float* src = (mat == 0) ? wq : (mat == 1) ? wk : wv;
    Wb[idx] = (__bf16)src[off];
}

// ---------------------------------------------------------------------------
// Kernel 3: Q = F Wq^T + bq ; K = F Wk^T + bk ; V^T (stored [D][P]).
// One 16x16 tile per wave. grid = (2048, 3), block = 256.
// ---------------------------------------------------------------------------
__global__ void k_qkv_gemm(const __bf16* __restrict__ Fb,
                           const __bf16* __restrict__ Wb,
                           const float* __restrict__ bq,
                           const float* __restrict__ bk,
                           const float* __restrict__ bv,
                           __bf16* __restrict__ Qb,
                           __bf16* __restrict__ Kb,
                           __bf16* __restrict__ Vt) {
    const int mat  = blockIdx.y;
    const int wave = threadIdx.x >> 5;
    const int lane = threadIdx.x & 31;
    const int tile = blockIdx.x * 8 + wave;      // 512*32 tiles
    const int tm   = tile >> 5;                  // M tile (rows of F)
    const int tn   = tile & 31;                  // N tile (output feature)
    const __bf16* W    = Wb + (size_t)mat * D_DIM * D_DIM;
    const float*  bias = (mat == 0) ? bq : (mat == 1) ? bk : bv;

    v8f c = {};
    #pragma unroll 4
    for (int k0 = 0; k0 < D_DIM; k0 += 32) {
        v16bf a = load_frag_a(Fb, D_DIM, tm * 16, k0, lane);
        v16bf b = load_frag_b(W,  D_DIM, tn * 16, k0, lane);
        c = wmma_bf16(a, b, c);
    }
    const int m_  = lane & 15;
    const int hi  = lane >> 4;
    const int col = tn * 16 + m_;
    const float bval = bias[col];
    if (mat < 2) {
        __bf16* out = (mat == 0) ? Qb : Kb;
        #pragma unroll
        for (int v = 0; v < 8; ++v) {
            const int row = tm * 16 + v + 8 * hi;
            out[(size_t)row * D_DIM + col] = (__bf16)(c[v] + bval);
        }
    } else {
        #pragma unroll
        for (int v = 0; v < 8; ++v) {
            const int row = tm * 16 + v + 8 * hi;          // chunk index p
            Vt[(size_t)col * P_DIM + row] = (__bf16)(c[v] + bval);
        }
    }
}

// ---------------------------------------------------------------------------
// Kernel 4: dist (to d_out) + row softmax stats (rowmax, rowsumexp).
// Block = 8 waves sharing a 64-row tile (4 row-subtiles), Q/F staged in LDS.
// Wave w sweeps column tiles [w*64, (w+1)*64); every K/F B-fragment feeds
// 8 WMMAs (4 row-subtiles x {QK^T, FF^T}) -> 4x less L2 streaming than a
// 16-row block.  Per-lane online (m,l); one cross-lane merge at the end.
// ---------------------------------------------------------------------------
__global__ void k_attn_pass1(const __bf16* __restrict__ Fb,
                             const __bf16* __restrict__ Qb,
                             const __bf16* __restrict__ Kb,
                             const float* __restrict__ sqn,
                             const float* __restrict__ lam_p,
                             float* __restrict__ dist_out,
                             float* __restrict__ rowmax,
                             float* __restrict__ rowsum) {
    const int wave = threadIdx.x >> 5;
    const int lane = threadIdx.x & 31;
    const int r0   = blockIdx.x * 64;
    const int m_   = lane & 15;
    const int hi   = lane >> 4;
    const float lam = lam_p[0];

    // Stage the block's Q and F row tiles (64x512 bf16 each, 64 KB each).
    __shared__ __bf16 qs[64 * D_DIM];
    __shared__ __bf16 fs[64 * D_DIM];
    {
        const uint4* gq = (const uint4*)(Qb + (size_t)r0 * D_DIM);
        const uint4* gf = (const uint4*)(Fb + (size_t)r0 * D_DIM);
        uint4* sq = (uint4*)qs;
        uint4* sf = (uint4*)fs;
        #pragma unroll
        for (int j = 0; j < 16; ++j) {
            const int idx = j * 256 + threadIdx.x;   // 4096 uint4 each
            sq[idx] = gq[idx];
            sf[idx] = gf[idx];
        }
    }
    __syncthreads();

    float sqr[4][8];
    #pragma unroll
    for (int rt = 0; rt < 4; ++rt)
        #pragma unroll
        for (int v = 0; v < 8; ++v)
            sqr[rt][v] = sqn[r0 + rt * 16 + v + 8 * hi];

    float mr[4][8], lr[4][8];
    #pragma unroll
    for (int rt = 0; rt < 4; ++rt)
        #pragma unroll
        for (int v = 0; v < 8; ++v) { mr[rt][v] = -3.0e38f; lr[rt][v] = 0.f; }

    for (int t = wave * 64; t < wave * 64 + 64; ++t) {
        const int n0 = t * 16;
        if (t + 1 < wave * 64 + 64) {
            const size_t nr = (size_t)(n0 + 16 + m_) * D_DIM;
            __builtin_prefetch(Kb + nr, 0, 1);
            __builtin_prefetch(Fb + nr, 0, 1);
        }
        v8f cq[4] = {{}, {}, {}, {}};
        v8f cf[4] = {{}, {}, {}, {}};
        #pragma unroll 2
        for (int k0 = 0; k0 < D_DIM; k0 += 32) {
            v16bf bk_ = load_frag_b(Kb, D_DIM, n0, k0, lane);
            v16bf bf_ = load_frag_b(Fb, D_DIM, n0, k0, lane);
            #pragma unroll
            for (int rt = 0; rt < 4; ++rt) {
                v16bf aq = load_frag_a(qs, D_DIM, rt * 16, k0, lane);
                cq[rt] = wmma_bf16(aq, bk_, cq[rt]);
                v16bf af = load_frag_a(fs, D_DIM, rt * 16, k0, lane);
                cf[rt] = wmma_bf16(af, bf_, cf[rt]);
            }
        }
        const float sqc = sqn[n0 + m_];
        #pragma unroll
        for (int rt = 0; rt < 4; ++rt) {
            #pragma unroll
            for (int v = 0; v < 8; ++v) {
                float sq = sqr[rt][v] + sqc - 2.0f * cf[rt][v];
                sq = fmaxf(sq, 0.0f);
                const float dv = __builtin_sqrtf(sq);
                const int row = r0 + rt * 16 + v + 8 * hi;
                dist_out[(size_t)row * P_DIM + n0 + m_] = dv;
                const float s = cq[rt][v] * SCALE_QK - lam * dv;
                const float mnew = fmaxf(mr[rt][v], s);
                lr[rt][v] = lr[rt][v] * __expf(mr[rt][v] - mnew)
                          + __expf(s - mnew);
                mr[rt][v] = mnew;
            }
        }
    }

    // Merge (m,l) across the 16 lanes of each half.
    #pragma unroll
    for (int msk = 1; msk <= 8; msk <<= 1) {
        #pragma unroll
        for (int rt = 0; rt < 4; ++rt) {
            #pragma unroll
            for (int v = 0; v < 8; ++v) {
                const float mo = __shfl_xor(mr[rt][v], msk, 32);
                const float lo = __shfl_xor(lr[rt][v], msk, 32);
                const float mn = fmaxf(mr[rt][v], mo);
                lr[rt][v] = lr[rt][v] * __expf(mr[rt][v] - mn)
                          + lo * __expf(mo - mn);
                mr[rt][v] = mn;
            }
        }
    }

    // Combine across the 8 waves.
    __shared__ float sm[8][64];
    __shared__ float sl[8][64];
    if (m_ == 0) {
        #pragma unroll
        for (int rt = 0; rt < 4; ++rt)
            #pragma unroll
            for (int v = 0; v < 8; ++v) {
                const int row = rt * 16 + v + 8 * hi;
                sm[wave][row] = mr[rt][v];
                sl[wave][row] = lr[rt][v];
            }
    }
    __syncthreads();
    if (threadIdx.x < 64) {
        const int row = threadIdx.x;
        float M = -3.0e38f;
        #pragma unroll
        for (int w = 0; w < 8; ++w) M = fmaxf(M, sm[w][row]);
        float L = 0.f;
        #pragma unroll
        for (int w = 0; w < 8; ++w) L += sl[w][row] * __expf(sm[w][row] - M);
        rowmax[r0 + row] = M;
        rowsum[r0 + row] = L;
    }
}

// ---------------------------------------------------------------------------
// Kernel 5: F_out = softmax(QK^T/sqrt(d) - lam*dist) @ V.
// Block = 8 waves, 64-row tile (Q staged in LDS), column super-steps of 128.
// Wave w computes the s/p subtile for columns [n0+16w, n0+16w+16) over all
// 4 row-subtiles, then accumulates the 64-column d-chunk [64w, 64w+64) of O
// (16 accumulators, unnormalized); every V B-fragment feeds 4 WMMAs.
// Normalization by 1/rowsum happens once at the final store.
// ---------------------------------------------------------------------------
__global__ void k_attn_pass2(const __bf16* __restrict__ Qb,
                             const __bf16* __restrict__ Kb,
                             const __bf16* __restrict__ Vt,
                             const float* __restrict__ lam_p,
                             const float* __restrict__ dist,
                             const float* __restrict__ rowmax,
                             const float* __restrict__ rowsum,
                             float* __restrict__ Fout) {
    const int wave = threadIdx.x >> 5;
    const int lane = threadIdx.x & 31;
    const int r0   = blockIdx.x * 64;
    const int m_   = lane & 15;
    const int hi   = lane >> 4;
    const float lam = lam_p[0];

    __shared__ __bf16 qs[64 * D_DIM];   // 64 KB Q tile
    __shared__ __bf16 pt[64][128];      // 16 KB probability tile
    {
        const uint4* gq = (const uint4*)(Qb + (size_t)r0 * D_DIM);
        uint4* sq = (uint4*)qs;
        #pragma unroll
        for (int j = 0; j < 16; ++j) {
            const int idx = j * 256 + threadIdx.x;
            sq[idx] = gq[idx];
        }
    }
    __syncthreads();

    float rm[4][8];
    #pragma unroll
    for (int rt = 0; rt < 4; ++rt)
        #pragma unroll
        for (int v = 0; v < 8; ++v)
            rm[rt][v] = rowmax[r0 + rt * 16 + v + 8 * hi];

    v8f o[4][4];
    #pragma unroll
    for (int rt = 0; rt < 4; ++rt)
        #pragma unroll
        for (int nb = 0; nb < 4; ++nb) o[rt][nb] = (v8f){};

    for (int n0 = 0; n0 < P_DIM; n0 += 128) {
        const int nc = n0 + wave * 16;
        // s subtiles for this wave's 16 columns, all 4 row-subtiles.
        v8f cqk[4] = {{}, {}, {}, {}};
        #pragma unroll 2
        for (int k0 = 0; k0 < D_DIM; k0 += 32) {
            v16bf bk_ = load_frag_b(Kb, D_DIM, nc, k0, lane);
            #pragma unroll
            for (int rt = 0; rt < 4; ++rt) {
                v16bf aq = load_frag_a(qs, D_DIM, rt * 16, k0, lane);
                cqk[rt] = wmma_bf16(aq, bk_, cqk[rt]);
            }
        }
        #pragma unroll
        for (int rt = 0; rt < 4; ++rt) {
            #pragma unroll
            for (int v = 0; v < 8; ++v) {
                const int row = r0 + rt * 16 + v + 8 * hi;
                const float dv = dist[(size_t)row * P_DIM + nc + m_];
                const float s  = cqk[rt][v] * SCALE_QK - lam * dv;
                const float p  = __expf(s - rm[rt][v]);   // unnormalized
                pt[rt * 16 + v + 8 * hi][wave * 16 + m_] = (__bf16)p;
            }
        }
        __syncthreads();
        // O[64 x 64-chunk] += P[64 x 128] @ V[128 x 64-chunk]
        #pragma unroll
        for (int kk = 0; kk < 128; kk += 32) {
            v16bf ap[4];
            #pragma unroll
            for (int rt = 0; rt < 4; ++rt) {
                const __bf16* pp = &pt[rt * 16 + m_][kk + hi * 8];
                bf16x8 lo = *(const bf16x8*)pp;
                bf16x8 h8 = *(const bf16x8*)(pp + 16);
                ap[rt] = __builtin_shufflevector(lo, h8,
                    0,1,2,3,4,5,6,7,8,9,10,11,12,13,14,15);
            }
            const int kglob = n0 + kk;
            #pragma unroll
            for (int nb = 0; nb < 4; ++nb) {
                v16bf bv_ = load_frag_b(Vt, P_DIM, wave * 64 + nb * 16,
                                        kglob, lane);
                #pragma unroll
                for (int rt = 0; rt < 4; ++rt)
                    o[rt][nb] = wmma_bf16(ap[rt], bv_, o[rt][nb]);
            }
        }
        __syncthreads();
    }

    #pragma unroll
    for (int rt = 0; rt < 4; ++rt) {
        #pragma unroll
        for (int v = 0; v < 8; ++v) {
            const int row = r0 + rt * 16 + v + 8 * hi;
            const float ril = 1.0f / rowsum[row];
            float* outr = Fout + (size_t)row * D_DIM + wave * 64 + m_;
            outr[0]  = o[rt][0][v] * ril;
            outr[16] = o[rt][1][v] * ril;
            outr[32] = o[rt][2][v] * ril;
            outr[48] = o[rt][3][v] * ril;
        }
    }
}

// ---------------------------------------------------------------------------
// Host-side launcher
// ---------------------------------------------------------------------------
extern "C" void kernel_launch(void* const* d_in, const int* in_sizes, int n_in,
                              void* d_out, int out_size, void* d_ws, size_t ws_size,
                              hipStream_t stream) {
    const float* F  = (const float*)d_in[0];
    const float* Wq = (const float*)d_in[1];
    const float* bq = (const float*)d_in[2];
    const float* Wk = (const float*)d_in[3];
    const float* bk = (const float*)d_in[4];
    const float* Wv = (const float*)d_in[5];
    const float* bv = (const float*)d_in[6];
    const float* lam = (const float*)d_in[7];

    float* Fout = (float*)d_out;                             // [P][D]
    float* dist = (float*)d_out + (size_t)P_DIM * D_DIM;     // [P][P]

    char* ws = (char*)d_ws;
    const size_t SZ_PD = (size_t)P_DIM * D_DIM * sizeof(__bf16);   // 8 MB
    __bf16* Fb = (__bf16*)(ws);
    __bf16* Qb = (__bf16*)(ws + SZ_PD);
    __bf16* Kb = (__bf16*)(ws + 2 * SZ_PD);
    __bf16* Vt = (__bf16*)(ws + 3 * SZ_PD);
    __bf16* Wb = (__bf16*)(ws + 4 * SZ_PD);                        // 1.5 MB
    float* sqn    = (float*)(ws + 4 * SZ_PD + 3 * (size_t)D_DIM * D_DIM * sizeof(__bf16));
    float* rowmax = sqn + P_DIM;
    float* rowsum = rowmax + P_DIM;

    // 1) F -> bf16 + squared norms
    k_prep_rows<<<P_DIM / 8, 256, 0, stream>>>(F, Fb, sqn);
    // 2) weights -> bf16
    k_prep_w<<<(3 * D_DIM * D_DIM) / 256, 256, 0, stream>>>(Wq, Wk, Wv, Wb);
    // 3) Q, K, V^T
    k_qkv_gemm<<<dim3(2048, 3), 256, 0, stream>>>(Fb, Wb, bq, bk, bv, Qb, Kb, Vt);
    // 4) dist + softmax row stats
    k_attn_pass1<<<P_DIM / 64, 256, 0, stream>>>(Fb, Qb, Kb, sqn, lam,
                                                 dist, rowmax, rowsum);
    // 5) attn @ V
    k_attn_pass2<<<P_DIM / 64, 256, 0, stream>>>(Qb, Kb, Vt, lam,
                                                 dist, rowmax, rowsum, Fout);
}
